// HGCL_47132971106886
// MI455X (gfx1250) — matrix operand order repaired and checked by reading, not compiled, not run on hardware
//
#include <hip/hip_runtime.h>
#include <hip/hip_bf16.h>

#define EPSF   1e-7f
#define MINN   1e-15f
#define MAXN   1000000.0f

typedef __attribute__((ext_vector_type(16))) _Float16 v16h;
typedef __attribute__((ext_vector_type(8)))  _Float16 v8h;
typedef __attribute__((ext_vector_type(8)))  float    v8f;

// ---------- wave helpers (wave32) ----------
__device__ __forceinline__ float wred(float v) {
    v += __shfl_xor(v, 16, 32);
    v += __shfl_xor(v, 8, 32);
    v += __shfl_xor(v, 4, 32);
    v += __shfl_xor(v, 2, 32);
    v += __shfl_xor(v, 1, 32);
    return v;  // all lanes hold the sum
}
__device__ __forceinline__ float arcosh_(float t) {
    return logf(t + sqrtf(t * t - 1.0f));
}
__device__ __forceinline__ float sigmoidf_(float z) {
    return 1.0f / (1.0f + expf(-z));
}

// ---------- setup: W -> half copy, ub = logmap0(proj(expmap0(proj_tan0(b)))) ----------
__global__ void hgcl_setup(const float* __restrict__ W, const float* __restrict__ b,
                           _Float16* __restrict__ Wh, float* __restrict__ ubv) {
    int tid = threadIdx.x;
    for (int i = tid; i < 64 * 64; i += 256) Wh[i] = (_Float16)W[i];
    if (tid < 32) {
        int j = tid;
        // proj_tan0: zero component 0
        float ba = (j == 0) ? 0.0f : b[j];
        float bb = b[j + 32];
        // expmap0
        float xn = fmaxf(sqrtf(wred(ba * ba + bb * bb)), MINN);
        float rr = sinhf(xn) / xn;
        float spa = ba * rr, spb = bb * rr;
        float sn2 = wred(spa * spa + spb * spb);       // spa==0 at j==0
        float x0  = sqrtf(fmaxf(1.0f + sn2, EPSF));    // proj
        // logmap0
        float yn = fmaxf(sqrtf(sn2), MINN);
        float sc = arcosh_(fmaxf(x0, 1.0f + EPSF)) / yn;
        ubv[j]      = (j == 0) ? 0.0f : spa * sc;
        ubv[j + 32] = spb * sc;
    }
}

// ---------- per-node pre: logmap0 -> layernorm -> expmap0 -> logmap0 -> half ----------
__global__ void hgcl_node_pre(const float* __restrict__ h,
                              const float* __restrict__ g, const float* __restrict__ be,
                              _Float16* __restrict__ Ah, int N) {
    int node = blockIdx.x * 8 + (threadIdx.x >> 5);
    int j = threadIdx.x & 31;
    if (node >= N) return;
    const float* hp = h + (size_t)node * 64;
    float ha = hp[j], hb = hp[j + 32];
    // logmap0(h)
    float yn = fmaxf(sqrtf(wred(((j == 0) ? 0.0f : ha * ha) + hb * hb)), MINN);
    float h0 = __shfl(ha, 0, 32);
    float sc = arcosh_(fmaxf(h0, 1.0f + EPSF)) / yn;
    float ta = (j == 0) ? 0.0f : ha * sc;
    float tb = hb * sc;
    // layernorm over 63 spatial components
    float mu = wred(ta + tb) * (1.0f / 63.0f);
    float da = (j == 0) ? 0.0f : (ta - mu);
    float db = tb - mu;
    float var = wred(da * da + db * db) * (1.0f / 63.0f);
    float rstd = rsqrtf(var + 1e-5f);
    float ya, yb;
    if (j == 0) ya = 0.0f; else ya = da * rstd * g[j - 1] + be[j - 1];
    yb = db * rstd * g[j + 31] + be[j + 31];
    // h1 = expmap0([0,y]); t2 = logmap0(h1)
    float xn = fmaxf(sqrtf(wred(ya * ya + yb * yb)), MINN);
    float rr = sinhf(xn) / xn;
    float spa = (j == 0) ? 0.0f : ya * rr;
    float spb = yb * rr;
    float sn2 = wred(spa * spa + spb * spb);
    float x0  = sqrtf(fmaxf(1.0f + sn2, EPSF));
    float yn2 = fmaxf(sqrtf(sn2), MINN);
    float sc2 = arcosh_(fmaxf(x0, 1.0f + EPSF)) / yn2;
    _Float16* ap = Ah + (size_t)node * 64;
    ap[j]      = (_Float16)((j == 0) ? 0.0f : spa * sc2);
    ap[j + 32] = (_Float16)(spb * sc2);
}

// ---------- WMMA GEMM: C[N,64] = A[N,64] * W^T ;  B[k,n] = Wh[n*64+k] ----------
__global__ __launch_bounds__(256) void hgcl_gemm(const _Float16* __restrict__ Ah,
                                                 const _Float16* __restrict__ Wh,
                                                 float* __restrict__ C, int N) {
    int wave = threadIdx.x >> 5;
    int lane = threadIdx.x & 31;
    int rowTile = blockIdx.x * 2 + (wave >> 2);
    int colTile = wave & 3;
    int m0 = rowTile * 16;
    if (m0 >= N) return;                       // wave-uniform; EXEC stays all-ones
    int lr = lane & 15, hi = lane >> 4;
    int m = m0 + lr;
    int n = colTile * 16 + lr;
    v8f acc = {};
#pragma unroll
    for (int s = 0; s < 2; ++s) {
        // A 16x32 f16 layout: halves 0..7 -> K = s*32 + hi*16 + i ; halves 8..15 -> K = s*32 + hi*16 + 8 + i
        v8h alo = *(const v8h*)(Ah + (size_t)m * 64 + s * 32 + hi * 16);
        v8h ahi = *(const v8h*)(Ah + (size_t)m * 64 + s * 32 + hi * 16 + 8);
        v16h a;
#pragma unroll
        for (int i = 0; i < 8; ++i) { a[i] = alo[i]; a[8 + i] = ahi[i]; }
        // B 32x16 f16 layout: lane covers column n, K = s*32 + hi*16 + (0..15) contiguous
        v16h bv = *(const v16h*)(Wh + (size_t)n * 64 + s * 32 + hi * 16);
        acc = __builtin_amdgcn_wmma_f32_16x16x32_f16(false, a, false, bv, (short)0, acc,
                                                     false, false);
    }
    // D layout: VGPR r -> M = r + 8*hi, N = lr
#pragma unroll
    for (int r2 = 0; r2 < 8; ++r2)
        C[(size_t)(m0 + r2 + hi * 8) * 64 + colTile * 16 + lr] = acc[r2];
}

// ---------- per-node mid: mv = proj(expmap0(.)); mobius_add with bias; h2, x_t ----------
__global__ void hgcl_node_mid(const float* __restrict__ mvt, const float* __restrict__ ubv,
                              float* __restrict__ h2o, float* __restrict__ xto, int N) {
    int node = blockIdx.x * 8 + (threadIdx.x >> 5);
    int j = threadIdx.x & 31;
    if (node >= N) return;
    const float* mp = mvt + (size_t)node * 64;
    float va = mp[j], vb = mp[j + 32];
    // mv = proj(expmap0(mvt))
    float xn = fmaxf(sqrtf(wred(((j == 0) ? 0.0f : va * va) + vb * vb)), MINN);
    float rr = sinhf(xn) / xn;
    float spa = (j == 0) ? 0.0f : va * rr;
    float spb = vb * rr;
    float sn2 = wred(spa * spa + spb * spb);
    float x0  = sqrtf(fmaxf(1.0f + sn2, EPSF));
    // ptransp0(mv, ub)
    float ua = ubv[j], ub2 = ubv[j + 32];     // ub[0] == 0
    float yn = fmaxf(sqrtf(sn2), MINN);
    float yua = spa / yn, yub = spb / yn;
    float alpha = wred(yua * ua + yub * ub2); // dot over spatial (yua==0 at j==0)
    float vva = (j == 0) ? -yn : (1.0f - x0) * yua;
    float vvb = (1.0f - x0) * yub;
    float wa = ua - alpha * vva;
    float wb = ub2 - alpha * vvb;
    // proj_tan(w, mv)
    float ux = wred(spa * wa + spb * wb);
    float w0 = ux / fmaxf(x0, EPSF);
    if (j == 0) wa = w0;
    // h2 = proj(expmap(w, mv))
    float md = wred(((j == 0) ? -wa * wa : wa * wa) + wb * wb);
    float normu = fminf(sqrtf(fmaxf(md, EPSF)), MAXN);
    float theta = fmaxf(normu, MINN);
    float ch = coshf(theta), sr = sinhf(theta) / theta;
    float za = ch * ((j == 0) ? x0 : spa) + sr * wa;
    float zb = ch * spb + sr * wb;
    float zs2 = wred(((j == 0) ? 0.0f : za * za) + zb * zb);
    float z0 = sqrtf(fmaxf(1.0f + zs2, EPSF));
    float h2a = (j == 0) ? z0 : za;
    float* hp = h2o + (size_t)node * 64;
    hp[j] = h2a; hp[j + 32] = zb;
    // x_t = logmap0(h2)
    float yn3 = fmaxf(sqrtf(zs2), MINN);
    float sc3 = arcosh_(fmaxf(z0, 1.0f + EPSF)) / yn3;
    float* xp = xto + (size_t)node * 64;
    xp[j]      = (j == 0) ? 0.0f : za * sc3;
    xp[j + 32] = zb * sc3;
}

// ---------- zero agg ----------
__global__ void hgcl_zero(float* __restrict__ p, int n) {
    int i = blockIdx.x * blockDim.x + threadIdx.x;
    int stride = gridDim.x * blockDim.x;
    for (; i < n; i += stride) p[i] = 0.0f;
}

// ---------- per-edge: x_local = logmap(h2[r], h2[c]); att; atomic scatter ----------
__global__ void hgcl_edge(const int* __restrict__ ei, const float* __restrict__ ea,
                          const float* __restrict__ emask,
                          const float* __restrict__ h2, const float* __restrict__ xt,
                          const float* __restrict__ aw, const float* __restrict__ ab,
                          float* __restrict__ agg, int E) {
    int e = blockIdx.x * 8 + (threadIdx.x >> 5);
    int j = threadIdx.x & 31;
    if (e >= E) return;
    int r = ei[e];
    int c = ei[E + e];
    const float* xr = h2 + (size_t)r * 64;
    const float* yc = h2 + (size_t)c * 64;
    float xa = xr[j], xb = xr[j + 32];
    float ya = yc[j], yb = yc[j + 32];
    // mdot(x,y) = sum - 2*x0*y0
    float md = wred(((j == 0) ? -xa * ya : xa * ya) + xb * yb);
    float xy = fminf(md + 1.0f, -EPSF) - 1.0f;
    float ua = ya + xy * xa;
    float ub2 = yb + xy * xb;
    float mu2 = wred(((j == 0) ? -ua * ua : ua * ua) + ub2 * ub2);
    float normu = fmaxf(sqrtf(fmaxf(mu2, EPSF)), MINN);
    float thd = fmaxf(-md, 1.0f + EPSF);
    float ach = arcosh_(thd);
    float dist = sqrtf(fminf(ach * ach, 50.0f));
    float s = dist / normu;
    float wa = s * ua, wb = s * ub2;
    // proj_tan(w, x)
    float ux = wred(((j == 0) ? 0.0f : xa * wa) + xb * wb);
    float x0 = __shfl(xa, 0, 32);
    float w0 = ux / fmaxf(x0, EPSF);
    if (j == 0) wa = w0;
    // attention: [x_t[r], x_t[c], edge_attr] . att_w + att_b
    const float* tr = xt + (size_t)r * 64;
    const float* tc = xt + (size_t)c * 64;
    float z = aw[j] * tr[j] + aw[j + 32] * tr[j + 32] +
              aw[64 + j] * tc[j] + aw[96 + j] * tc[j + 32];
    if (j == 0) z += aw[128] * ea[(size_t)e * 2] + aw[129] * ea[(size_t)e * 2 + 1] + ab[0];
    z = wred(z);
    float att = sigmoidf_(z) * emask[e];
    atomicAdd(&agg[(size_t)r * 64 + j], wa * att);
    atomicAdd(&agg[(size_t)r * 64 + j + 32], wb * att);
}

// ---------- per-node post: h3 = proj(expmap(agg, h2)); silu(logmap0); expmap0; proj ----------
__global__ void hgcl_node_post(const float* __restrict__ agg, const float* __restrict__ nmask,
                               const float* __restrict__ h2, float* __restrict__ out, int N) {
    int node = blockIdx.x * 8 + (threadIdx.x >> 5);
    int j = threadIdx.x & 31;
    if (node >= N) return;
    float m = nmask[node];
    const float* ap = agg + (size_t)node * 64;
    const float* hp = h2 + (size_t)node * 64;
    float ua = ap[j] * m, ub2 = ap[j + 32] * m;
    float xa = hp[j], xb = hp[j + 32];
    // expmap(u, h2)
    float md = wred(((j == 0) ? -ua * ua : ua * ua) + ub2 * ub2);
    float normu = fminf(sqrtf(fmaxf(md, EPSF)), MAXN);
    float theta = fmaxf(normu, MINN);
    float ch = coshf(theta), sr = sinhf(theta) / theta;
    float za = ch * xa + sr * ua;
    float zb = ch * xb + sr * ub2;
    // proj
    float zs2 = wred(((j == 0) ? 0.0f : za * za) + zb * zb);
    float z0 = sqrtf(fmaxf(1.0f + zs2, EPSF));
    // logmap0(h3)
    float yn = fmaxf(sqrtf(zs2), MINN);
    float sc = arcosh_(fmaxf(z0, 1.0f + EPSF)) / yn;
    float la = (j == 0) ? 0.0f : za * sc;
    float lb = zb * sc;
    // silu, proj_tan0
    float sa = (j == 0) ? 0.0f : la * sigmoidf_(la);
    float sb = lb * sigmoidf_(lb);
    // expmap0 + proj
    float xn = fmaxf(sqrtf(wred(sa * sa + sb * sb)), MINN);
    float rr = sinhf(xn) / xn;
    float oa = (j == 0) ? 0.0f : sa * rr;
    float ob = sb * rr;
    float on2 = wred(oa * oa + ob * ob);
    float o0 = sqrtf(fmaxf(1.0f + on2, EPSF));
    float* op = out + (size_t)node * 64;
    op[j]      = (j == 0) ? o0 : oa;
    op[j + 32] = ob;
}

extern "C" void kernel_launch(void* const* d_in, const int* in_sizes, int n_in,
                              void* d_out, int out_size, void* d_ws, size_t ws_size,
                              hipStream_t stream) {
    const float* h       = (const float*)d_in[0];
    const int*   ei      = (const int*)d_in[1];
    const float* ea      = (const float*)d_in[2];
    const float* nmask   = (const float*)d_in[3];
    const float* emask   = (const float*)d_in[4];
    const float* g       = (const float*)d_in[5];
    const float* be      = (const float*)d_in[6];
    const float* W       = (const float*)d_in[7];
    const float* b       = (const float*)d_in[8];
    const float* aw      = (const float*)d_in[9];
    const float* ab      = (const float*)d_in[10];
    float* out = (float*)d_out;

    const int N = in_sizes[0] / 64;
    const int E = in_sizes[1] / 2;

    // workspace carving (256-B aligned)
    char* ws = (char*)d_ws;
    size_t off = 0;
    auto carve = [&](size_t bytes) -> char* {
        char* p = ws + off;
        off = (off + bytes + 255) & ~(size_t)255;
        return p;
    };
    float*    ubv = (float*)   carve(64 * sizeof(float));
    _Float16* Wh  = (_Float16*)carve(64 * 64 * sizeof(_Float16));
    _Float16* Ah  = (_Float16*)carve((size_t)N * 64 * sizeof(_Float16));
    float*    mvt = (float*)   carve((size_t)N * 64 * sizeof(float));
    float*    h2  = (float*)   carve((size_t)N * 64 * sizeof(float));
    float*    xt  = (float*)   carve((size_t)N * 64 * sizeof(float));
    float*    agg = (float*)   carve((size_t)N * 64 * sizeof(float));

    const int nodeBlocks = (N + 7) / 8;
    const int edgeBlocks = (E + 7) / 8;
    const int rowTiles   = (N + 15) / 16;
    const int gemmBlocks = (rowTiles + 1) / 2;

    hipLaunchKernelGGL(hgcl_setup, dim3(1), dim3(256), 0, stream, W, b, Wh, ubv);
    hipLaunchKernelGGL(hgcl_node_pre, dim3(nodeBlocks), dim3(256), 0, stream, h, g, be, Ah, N);
    hipLaunchKernelGGL(hgcl_gemm, dim3(gemmBlocks), dim3(256), 0, stream, Ah, Wh, mvt, N);
    hipLaunchKernelGGL(hgcl_node_mid, dim3(nodeBlocks), dim3(256), 0, stream, mvt, ubv, h2, xt, N);
    hipLaunchKernelGGL(hgcl_zero, dim3(2048), dim3(256), 0, stream, agg, N * 64);
    hipLaunchKernelGGL(hgcl_edge, dim3(edgeBlocks), dim3(256), 0, stream,
                       ei, ea, emask, h2, xt, aw, ab, agg, E);
    hipLaunchKernelGGL(hgcl_node_post, dim3(nodeBlocks), dim3(256), 0, stream,
                       agg, nmask, h2, out, N);
}
